// MultiHeadSelfAttentionWithPooling_70042326663965
// MI455X (gfx1250) — matrix operand order repaired and checked by reading, not compiled, not run on hardware
//
#include <hip/hip_runtime.h>
#include <hip/hip_bf16.h>
#include <stdint.h>

typedef __bf16 bf16;
typedef __attribute__((ext_vector_type(8)))  __bf16 v8bf;
typedef __attribute__((ext_vector_type(16))) __bf16 v16bf;
typedef __attribute__((ext_vector_type(8)))  float  v8f;

#define EMBED 1024
#define SEQ   4096
#define BATCH 4
#define HEADS 16
#define HD    64
#define NH    64      /* BATCH*HEADS */
#define S2    2048    /* SEQ/2 */
#define S4    1024    /* SEQ/4 */

// async copy of 32 contiguous bytes global -> LDS (no VGPR round trip).
// INST_OFFSET is added to BOTH the global and LDS addresses (ISA 08 §4.4).
__device__ __forceinline__ void async_copy_b32x(const bf16* g, const bf16* l) {
  uint32_t lo = (uint32_t)(uintptr_t)l;            // LDS byte offset (low bits)
  uint64_t ga = (uint64_t)(uintptr_t)g;
  asm volatile("global_load_async_to_lds_b128 %0, %1, off"
               :: "v"(lo), "v"(ga) : "memory");
  asm volatile("global_load_async_to_lds_b128 %0, %1, off offset:16"
               :: "v"(lo), "v"(ga) : "memory");
}

// ---------------------------------------------------------------- conversions
__global__ void cvt_f32_to_bf16(const float* __restrict__ in,
                                bf16* __restrict__ out, int n) {
  int i = blockIdx.x * blockDim.x + threadIdx.x;
  if (i < n) out[i] = (bf16)in[i];
}

// -------------------------------------------- head split + seq maxpool (x2)
// in: [BATCH, SEQ, EMBED] bf16   out: [NH, S2, HD] bf16 (optionally scaled)
__global__ void pool_heads(const bf16* __restrict__ in,
                           bf16* __restrict__ out, float scale) {
  int i = blockIdx.x * blockDim.x + threadIdx.x;   // over NH*S2*HD = 2^23
  int d = i & (HD - 1);
  int t = (i >> 6) & (S2 - 1);
  int n = i >> 17;                                  // / (S2*HD)
  int b = n >> 4, h = n & 15;
  const bf16* p = in + ((size_t)(b * SEQ + 2 * t) * EMBED) + h * HD + d;
  float a0 = (float)p[0];
  float a1 = (float)p[EMBED];
  out[i] = (bf16)(fmaxf(a0, a1) * scale);
}

// ------------------------------------------------------- tiled WMMA bf16 GEMM
// C[M,N] = A[M,K] @ B[K,N] + bias[N].  128 threads = 4 waves, 64x64 block tile.
// Double-buffered LDS staging via async global->LDS (ASYNCcnt).
#define ASTR 40   /* LDS row stride (elems) for 64x32 A tile : 80B, 16B-aligned */
#define BSTR 72   /* LDS row stride (elems) for 32x64 B tile : 144B             */

__global__ __launch_bounds__(128) void gemm_wmma_bf16(
    const bf16* __restrict__ A, const bf16* __restrict__ Bm,
    const float* __restrict__ bias, void* __restrict__ Cout,
    int M, int N, int K, int out_f32) {
  __shared__ __align__(16) bf16 As[2][64 * ASTR];
  __shared__ __align__(16) bf16 Bs[2][32 * BSTR];

  const int tid = threadIdx.x;
  const int w  = tid >> 5;          // wave 0..3 -> 16-row slice
  const int l  = tid & 31;
  const int lr = l & 15;
  const int hi = l >> 4;
  const int m0 = blockIdx.x * 64;
  const int n0 = blockIdx.y * 64;

  v8f acc[4];
#pragma unroll
  for (int nt = 0; nt < 4; ++nt) acc[nt] = (v8f){0.f,0.f,0.f,0.f,0.f,0.f,0.f,0.f};

  const int arow = tid >> 1, acol = (tid & 1) * 16;  // A: 64 rows x 32 cols
  const int brow = tid >> 2, bcol = (tid & 3) * 16;  // B: 32 rows x 64 cols

  // issue one k-tile (4 async b128 ops per thread) into LDS buffer `buf`
  auto issue = [&](int kb, int buf) {
    const bf16* ag = A + (size_t)(m0 + arow) * K + kb + acol;
    async_copy_b32x(ag, &As[buf][arow * ASTR + acol]);
    const bf16* bg = Bm + (size_t)(kb + brow) * N + n0 + bcol;
    async_copy_b32x(bg, &Bs[buf][brow * BSTR + bcol]);
  };

  const int T = K >> 5;
  issue(0, 0);
  for (int t = 0; t < T; ++t) {
    const int cur = t & 1;
    if (t + 1 < T) {
      issue((t + 1) << 5, cur ^ 1);
      // 4 newest ops may stay in flight; the 4 for tile `cur` are complete
      asm volatile("s_wait_asynccnt 0x4" ::: "memory");
    } else {
      asm volatile("s_wait_asynccnt 0x0" ::: "memory");
    }
    __syncthreads();

    // A fragment (16x32), ISA 16-bit A layout: lane<16 -> K {0..7,16..23}
    v16bf af;
    {
      const int row = w * 16 + lr;
      v8bf lo = *(const v8bf*)&As[cur][row * ASTR + hi * 8];
      v8bf hb = *(const v8bf*)&As[cur][row * ASTR + 16 + hi * 8];
#pragma unroll
      for (int j = 0; j < 8; ++j) { af[j] = lo[j]; af[8 + j] = hb[j]; }
    }
#pragma unroll
    for (int nt = 0; nt < 4; ++nt) {
      // B fragment (32x16): lane = N col, vector index = K (table order)
      v16bf bfx;
#pragma unroll
      for (int j = 0; j < 8; ++j) {
        bfx[j]     = Bs[cur][(hi * 8 + j) * BSTR + nt * 16 + lr];
        bfx[8 + j] = Bs[cur][(16 + hi * 8 + j) * BSTR + nt * 16 + lr];
      }
      acc[nt] = __builtin_amdgcn_wmma_f32_16x16x32_bf16(
          false, af, false, bfx, (short)0, acc[nt], false, false);
    }
    __syncthreads();   // all waves done with `cur` before it is re-filled
  }

#pragma unroll
  for (int nt = 0; nt < 4; ++nt) {
    const int col = n0 + nt * 16 + lr;
    const float bv = bias[col];
#pragma unroll
    for (int r = 0; r < 8; ++r) {
      const int row = m0 + w * 16 + r + 8 * hi;   // ISA C layout
      float v = acc[nt][r] + bv;
      if (out_f32) ((float*)Cout)[(size_t)row * N + col] = v;
      else         ((bf16*)Cout)[(size_t)row * N + col]  = (bf16)v;
    }
  }
}

// ----------------------------------------------------------- flash attention
// grid: (S2/64 q-blocks, NH heads).  4 waves x 16 query rows.
// Qp pre-scaled by 1/sqrt(HD).  Epilogue fuses seq maxpool (2048 -> 1024).
__global__ __launch_bounds__(128) void flash_attn(
    const bf16* __restrict__ Qp, const bf16* __restrict__ Kp,
    const bf16* __restrict__ Vp, bf16* __restrict__ Op) {
  __shared__ __align__(16) bf16 Kts[64 * ASTR];       // K^T tile: [hd 0..63][key 0..31]
  __shared__ __align__(16) bf16 Vs[32 * BSTR];        // V tile:   [key 0..31][hd 0..63]
  __shared__ __align__(16) bf16 Pls[4][16 * ASTR];    // per-wave P (16x32) bounce

  const int tid = threadIdx.x;
  const int w  = tid >> 5, l = tid & 31, lr = l & 15, hi = l >> 4;
  const int n  = blockIdx.y;
  const int q0 = blockIdx.x * 64;

  const bf16* Qn = Qp + (size_t)n * S2 * HD;
  const bf16* Kn = Kp + (size_t)n * S2 * HD;
  const bf16* Vn = Vp + (size_t)n * S2 * HD;

  // Q fragments for this wave's 16 rows (hd split into two K=32 chunks)
  v16bf qf[2];
  {
    const bf16* qrow = Qn + (size_t)(q0 + w * 16 + lr) * HD;
#pragma unroll
    for (int c = 0; c < 2; ++c) {
      v8bf lo = *(const v8bf*)(qrow + c * 32 + hi * 8);
      v8bf hb = *(const v8bf*)(qrow + c * 32 + 16 + hi * 8);
#pragma unroll
      for (int j = 0; j < 8; ++j) { qf[c][j] = lo[j]; qf[c][8 + j] = hb[j]; }
    }
  }

  v8f acc[4];
#pragma unroll
  for (int nt = 0; nt < 4; ++nt) acc[nt] = (v8f){0.f,0.f,0.f,0.f,0.f,0.f,0.f,0.f};
  float mrow[8], lsum[8];
#pragma unroll
  for (int r = 0; r < 8; ++r) { mrow[r] = -3.0e38f; lsum[r] = 0.f; }

  const int krow = tid >> 2;         // 0..31
  const int kcg  = (tid & 3) * 16;   // 0,16,32,48

  for (int kb = 0; kb < S2; kb += 32) {
    __syncthreads();
    {   // V tile: async straight copy; K tile: VGPR transpose into LDS
      const bf16* vg = Vn + (size_t)(kb + krow) * HD + kcg;
      async_copy_b32x(vg, &Vs[krow * BSTR + kcg]);
      const bf16* kg = Kn + (size_t)(kb + krow) * HD + kcg;
      v8bf k0 = *(const v8bf*)kg;
      v8bf k1 = *(const v8bf*)(kg + 8);
#pragma unroll
      for (int j = 0; j < 8; ++j) {
        Kts[(kcg + j) * ASTR + krow]     = k0[j];
        Kts[(kcg + 8 + j) * ASTR + krow] = k1[j];
      }
    }
    asm volatile("s_wait_asynccnt 0x0" ::: "memory");
    __syncthreads();

    // S = Q @ K^T : two 16x16 score tiles (32 keys)
    v8f sc[2];
#pragma unroll
    for (int kt = 0; kt < 2; ++kt) {
      v8f s = (v8f){0.f,0.f,0.f,0.f,0.f,0.f,0.f,0.f};
#pragma unroll
      for (int c = 0; c < 2; ++c) {
        v16bf bfx;
#pragma unroll
        for (int j = 0; j < 8; ++j) {
          bfx[j]     = Kts[(c * 32 + hi * 8 + j) * ASTR + kt * 16 + lr];
          bfx[8 + j] = Kts[(c * 32 + 16 + hi * 8 + j) * ASTR + kt * 16 + lr];
        }
        s = __builtin_amdgcn_wmma_f32_16x16x32_bf16(
            false, qf[c], false, bfx, (short)0, s, false, false);
      }
      sc[kt] = s;
    }

    // online softmax (rows live across 16-lane halves of the C layout)
    float corr[8];
#pragma unroll
    for (int r = 0; r < 8; ++r) {
      float v = fmaxf(sc[0][r], sc[1][r]);
#pragma unroll
      for (int off = 8; off >= 1; off >>= 1) v = fmaxf(v, __shfl_xor(v, off, 32));
      float mn = fmaxf(mrow[r], v);
      float p0 = __expf(sc[0][r] - mn);
      float p1 = __expf(sc[1][r] - mn);
      float rs = p0 + p1;
#pragma unroll
      for (int off = 8; off >= 1; off >>= 1) rs += __shfl_xor(rs, off, 32);
      float c0 = __expf(mrow[r] - mn);
      lsum[r] = lsum[r] * c0 + rs;
      mrow[r] = mn;
      corr[r] = c0;
      Pls[w][(r + 8 * hi) * ASTR + lr]      = (bf16)p0;   // C-layout -> LDS
      Pls[w][(r + 8 * hi) * ASTR + 16 + lr] = (bf16)p1;
    }
#pragma unroll
    for (int nt = 0; nt < 4; ++nt)
#pragma unroll
      for (int r = 0; r < 8; ++r) acc[nt][r] *= corr[r];

    __syncthreads();   // make P visible across lanes; Kts reads done

    // P (16x32) as A fragment, V tiles as B fragments; O += P @ V
    v16bf pf;
    {
      v8bf lo = *(const v8bf*)&Pls[w][lr * ASTR + hi * 8];
      v8bf hb = *(const v8bf*)&Pls[w][lr * ASTR + 16 + hi * 8];
#pragma unroll
      for (int j = 0; j < 8; ++j) { pf[j] = lo[j]; pf[8 + j] = hb[j]; }
    }
#pragma unroll
    for (int nt = 0; nt < 4; ++nt) {
      v16bf vf;
#pragma unroll
      for (int j = 0; j < 8; ++j) {
        vf[j]     = Vs[(hi * 8 + j) * BSTR + nt * 16 + lr];
        vf[8 + j] = Vs[(16 + hi * 8 + j) * BSTR + nt * 16 + lr];
      }
      acc[nt] = __builtin_amdgcn_wmma_f32_16x16x32_bf16(
          false, pf, false, vf, (short)0, acc[nt], false, false);
    }
  }

  // normalize, fused seq-maxpool (adjacent rows = adjacent acc VGPRs), store
  float inv[8];
#pragma unroll
  for (int r = 0; r < 8; ++r) inv[r] = 1.f / lsum[r];
#pragma unroll
  for (int nt = 0; nt < 4; ++nt) {
#pragma unroll
    for (int j = 0; j < 4; ++j) {
      float v = fmaxf(acc[nt][2 * j] * inv[2 * j],
                      acc[nt][2 * j + 1] * inv[2 * j + 1]);
      int prow = (q0 >> 1) + w * 8 + hi * 4 + j;
      int col  = nt * 16 + lr;
      Op[((size_t)n * S4 + prow) * HD + col] = (bf16)v;
    }
  }
}

// ---------------------------------------------------------------- launcher
extern "C" void kernel_launch(void* const* d_in, const int* in_sizes, int n_in,
                              void* d_out, int out_size, void* d_ws, size_t ws_size,
                              hipStream_t stream) {
  const float* x  = (const float*)d_in[0];
  const float* Wq = (const float*)d_in[1];
  const float* bq = (const float*)d_in[2];
  const float* Wk = (const float*)d_in[3];
  const float* bk = (const float*)d_in[4];
  const float* Wv = (const float*)d_in[5];
  const float* bv = (const float*)d_in[6];
  const float* Wo = (const float*)d_in[7];
  const float* bo = (const float*)d_in[8];
  float* out = (float*)d_out;
  char*  ws  = (char*)d_ws;

  // workspace layout (bytes)
  constexpr size_t SZ_X   = (size_t)BATCH * SEQ * EMBED * 2;   // 32 MB
  constexpr size_t SZ_W   = (size_t)EMBED * EMBED * 2;         // 2 MB
  constexpr size_t SZ_P   = (size_t)NH * S2 * HD * 2;          // 16 MB
  bf16* xb  = (bf16*)(ws);
  bf16* Wqb = (bf16*)(ws + SZ_X);
  bf16* Wkb = (bf16*)(ws + SZ_X + SZ_W);
  bf16* Wvb = (bf16*)(ws + SZ_X + 2 * SZ_W);
  bf16* Wob = (bf16*)(ws + SZ_X + 3 * SZ_W);
  bf16* Qe  = (bf16*)(ws + SZ_X + 4 * SZ_W);
  bf16* Ke  = (bf16*)(ws + 2 * SZ_X + 4 * SZ_W);
  bf16* Ve  = (bf16*)(ws + 3 * SZ_X + 4 * SZ_W);
  bf16* Qp  = (bf16*)(ws + 4 * SZ_X + 4 * SZ_W);
  bf16* Kp  = (bf16*)(ws + 4 * SZ_X + 4 * SZ_W + SZ_P);
  bf16* Vp  = (bf16*)(ws + 4 * SZ_X + 4 * SZ_W + 2 * SZ_P);
  bf16* Opool = (bf16*)(ws + 4 * SZ_X + 4 * SZ_W + 3 * SZ_P);
  (void)ws_size; (void)n_in; (void)in_sizes; (void)out_size;

  const int nx = BATCH * SEQ * EMBED;   // 16,777,216
  const int nw = EMBED * EMBED;         // 1,048,576
  cvt_f32_to_bf16<<<nx / 256, 256, 0, stream>>>(x,  xb,  nx);
  cvt_f32_to_bf16<<<nw / 256, 256, 0, stream>>>(Wq, Wqb, nw);
  cvt_f32_to_bf16<<<nw / 256, 256, 0, stream>>>(Wk, Wkb, nw);
  cvt_f32_to_bf16<<<nw / 256, 256, 0, stream>>>(Wv, Wvb, nw);
  cvt_f32_to_bf16<<<nw / 256, 256, 0, stream>>>(Wo, Wob, nw);

  // QKV projections: [16384,1024] @ [1024,1024] + bias -> bf16
  dim3 gqkv((BATCH * SEQ) / 64, EMBED / 64);     // (256,16)
  gemm_wmma_bf16<<<gqkv, 128, 0, stream>>>(xb, Wqb, bq, Qe,
                                           BATCH * SEQ, EMBED, EMBED, 0);
  gemm_wmma_bf16<<<gqkv, 128, 0, stream>>>(xb, Wkb, bk, Ke,
                                           BATCH * SEQ, EMBED, EMBED, 0);
  gemm_wmma_bf16<<<gqkv, 128, 0, stream>>>(xb, Wvb, bv, Ve,
                                           BATCH * SEQ, EMBED, EMBED, 0);

  // head split + maxpool (Q pre-scaled by 1/sqrt(HD) = 0.125)
  const int np = NH * S2 * HD;          // 8,388,608
  pool_heads<<<np / 256, 256, 0, stream>>>(Qe, Qp, 0.125f);
  pool_heads<<<np / 256, 256, 0, stream>>>(Ke, Kp, 1.0f);
  pool_heads<<<np / 256, 256, 0, stream>>>(Ve, Vp, 1.0f);

  // flash attention + fused output maxpool -> [NH, S4, HD] bf16
  flash_attn<<<dim3(S2 / 64, NH), 128, 0, stream>>>(Qp, Kp, Vp, Opool);

  // final projection: flat [4096,1024] @ Wo + bo -> f32 d_out
  dim3 gout((BATCH * S4) / 64, EMBED / 64);      // (64,16)
  gemm_wmma_bf16<<<gout, 128, 0, stream>>>(Opool, Wob, bo, out,
                                           BATCH * S4, EMBED, EMBED, 1);
}
